// HeteroGNN_45372034515629
// MI455X (gfx1250) — compile-verified, matrix operand order checked by reading, not run on hardware
//
#include <hip/hip_runtime.h>
#include <hip/hip_bf16.h>
#include <math.h>

#define HID 64
#define CDIV(a,b) (((a)+(b)-1)/(b))

typedef __attribute__((ext_vector_type(16))) _Float16 v16h;
typedef __attribute__((ext_vector_type(8)))  float    v8f;

__device__ __forceinline__ float leakyf(float x, float s) { return x > 0.f ? x : x * s; }

__device__ __forceinline__ void atomicMaxF(float* addr, float val) {
    unsigned int* ua = (unsigned int*)addr;
    unsigned int cur = __float_as_uint(*addr);
    while (__uint_as_float(cur) < val) {
        unsigned int prev = atomicCAS(ua, cur, __float_as_uint(val));
        if (prev == cur) break;
        cur = prev;
    }
}

// ---------------------------------------------------------------------------
// Y[N x 64] = leaky( X[N x K] @ W[K x 64] + bias , slope )   (slope=1 -> id)
// K is compile-time (16/32/64) so all fragment addressing folds to constants.
// One wave computes a 16x64 stripe via NKB*4 x v_wmma_f32_16x16x32_f16.
// A fragment (16-bit 16x32): lane L -> M=L&15, half=L>>4;
//   elems 0..7  : K = kb*32 + half*8 + j      (contiguous 32B -> b128 loads)
//   elems 8..15 : K = kb*32 + 16 + half*8 + j
// B fragments are pre-swizzled into LDS once per block (W shared by all waves)
// so each wave reads a whole v16h as 2x ds_load_b128.
// C/D: lane L -> N=L&15, VGPR r -> M = r + 8*(L>>4).
// ---------------------------------------------------------------------------
template<int K>
__global__ __launch_bounds__(256) void k_wmma_linear(
    const float* __restrict__ X, const float* __restrict__ W,
    const float* __restrict__ bias, float slope,
    float* __restrict__ Y, int N)
{
    constexpr int NKB = (K + 31) / 32;
    __shared__ __align__(32) _Float16 shB[NKB][4][32][16];

    // cooperative pre-swizzle of W into WMMA B-fragment layout
    for (int idx = threadIdx.x; idx < NKB * 4 * 32 * 16; idx += 256) {
        const int j   = idx & 15;
        const int ln  = (idx >> 4) & 31;
        const int g   = (idx >> 9) & 3;
        const int kb  = idx >> 11;
        const int hfl = ln >> 4;
        const int col = g * 16 + (ln & 15);
        const int k   = (j < 8) ? (kb * 32 + hfl * 8 + j)
                                : (kb * 32 + 16 + hfl * 8 + (j - 8));
        shB[kb][g][ln][j] = (k < K) ? (_Float16)W[k * HID + col] : (_Float16)0.f;
    }
    __syncthreads();

    const int wave = threadIdx.x >> 5;
    const int lane = threadIdx.x & 31;
    const int m    = lane & 15;
    const int hf   = lane >> 4;
    const int rowBase = blockIdx.x * 128 + wave * 16;
    if (rowBase >= N) return;                         // wave-uniform exit

    // clamp load row into range: garbage rows computed but never stored
    const int rowc = min(rowBase + m, N - 1);
    const float* __restrict__ Xrow = X + (long)rowc * K;

    v8f acc[4] = {};

#pragma unroll
    for (int kb = 0; kb < NKB; ++kb) {
        v16h a;
        {   // chunk 0: k in [kb*32 + hf*8, +8)  -- always valid (K mult of 16)
            const float4 x0 = *(const float4*)(Xrow + kb * 32 + hf * 8);
            const float4 x1 = *(const float4*)(Xrow + kb * 32 + hf * 8 + 4);
            a[0] = (_Float16)x0.x; a[1] = (_Float16)x0.y;
            a[2] = (_Float16)x0.z; a[3] = (_Float16)x0.w;
            a[4] = (_Float16)x1.x; a[5] = (_Float16)x1.y;
            a[6] = (_Float16)x1.z; a[7] = (_Float16)x1.w;
        }
        if ((kb * 32 + 32) <= K) {   // compile-time after unroll
            const float4 x2 = *(const float4*)(Xrow + kb * 32 + 16 + hf * 8);
            const float4 x3 = *(const float4*)(Xrow + kb * 32 + 16 + hf * 8 + 4);
            a[8]  = (_Float16)x2.x; a[9]  = (_Float16)x2.y;
            a[10] = (_Float16)x2.z; a[11] = (_Float16)x2.w;
            a[12] = (_Float16)x3.x; a[13] = (_Float16)x3.y;
            a[14] = (_Float16)x3.z; a[15] = (_Float16)x3.w;
        } else {
#pragma unroll
            for (int j = 8; j < 16; ++j) a[j] = (_Float16)0.f;
        }
#pragma unroll
        for (int g = 0; g < 4; ++g) {
            const v16h b = *(const v16h*)(&shB[kb][g][lane][0]);
            acc[g] = __builtin_amdgcn_wmma_f32_16x16x32_f16(
                false, a, false, b, (short)0, acc[g], false, false);
        }
    }

    // epilogue: bias + leaky + store.  fullTile is wave-uniform, so the
    // common case takes an unconditional straight-line store path.
    float bv[4];
#pragma unroll
    for (int g = 0; g < 4; ++g) bv[g] = bias ? bias[g * 16 + m] : 0.f;

    float* __restrict__ Yrow = Y + (long)(rowBase + 8 * hf) * HID + m;
    if ((rowBase + 16) <= N) {
#pragma unroll
        for (int r = 0; r < 8; ++r) {
#pragma unroll
            for (int g = 0; g < 4; ++g) {
                Yrow[(long)r * HID + g * 16] = leakyf(acc[g][r] + bv[g], slope);
            }
        }
    } else {
#pragma unroll
        for (int r = 0; r < 8; ++r) {
            if (rowBase + r + 8 * hf < N) {
#pragma unroll
                for (int g = 0; g < 4; ++g) {
                    Yrow[(long)r * HID + g * 16] = leakyf(acc[g][r] + bv[g], slope);
                }
            }
        }
    }
}

// ---------------------------------------------------------------------------
__global__ void k_fill(float* p, float v, int n) {
    int i = blockIdx.x * blockDim.x + threadIdx.x;
    if (i < n) p[i] = v;
}

// out[i*64+j] = bias[j]   (add=0)   or   out += bias   (add=1)
__global__ void k_bias_rows(float* out, const float* __restrict__ bias, int n, int add) {
    int i = blockIdx.x * blockDim.x + threadIdx.x;
    if (i < n * HID) {
        float b = bias[i & 63];
        out[i] = add ? (out[i] + b) : b;
    }
}

// out[i] = dot(H[i, 0:64], a)
__global__ void k_row_dot(const float* __restrict__ H, const float* __restrict__ a,
                          float* __restrict__ out, int n) {
    int i = blockIdx.x * blockDim.x + threadIdx.x;
    if (i >= n) return;
    const float4* r  = (const float4*)(H + (long)i * HID);
    const float4* av = (const float4*)a;
    float acc = 0.f;
#pragma unroll
    for (int t = 0; t < 16; ++t) {
        float4 x = r[t], y = av[t];
        acc += x.x * y.x + x.y * y.y + x.z * y.z + x.w * y.w;
    }
    out[i] = acc;
}

// logit[e] = leaky( dsrc[src] + ddst[dst] + he[e].aedge , 0.2 );  m[dst] = max
__global__ void k_edge_logits(const float* __restrict__ he, const float* __restrict__ aedge,
                              const float* __restrict__ dsrc, const float* __restrict__ ddst,
                              const int* __restrict__ src, const int* __restrict__ dst,
                              float* __restrict__ logit, float* __restrict__ m, int E) {
    int e = blockIdx.x * blockDim.x + threadIdx.x;
    if (e >= E) return;
    const float4* r  = (const float4*)(he + (long)e * HID);
    const float4* av = (const float4*)aedge;
    float acc = 0.f;
#pragma unroll
    for (int t = 0; t < 16; ++t) {
        float4 x = r[t], y = av[t];
        acc += x.x * y.x + x.y * y.y + x.z * y.z + x.w * y.w;
    }
    int d = dst[e];
    float a = leakyf(dsrc[src[e]] + ddst[d] + acc, 0.2f);
    logit[e] = a;
    atomicMaxF(&m[d], a);
}

// self-loop logits for the homo convs (src=dst=i, he = mean edge feature)
__global__ void k_loop_logits(const float* __restrict__ dsrc, const float* __restrict__ ddst,
                              const float* __restrict__ cptr,
                              float* __restrict__ loopl, float* __restrict__ m, int n) {
    int i = blockIdx.x * blockDim.x + threadIdx.x;
    if (i >= n) return;
    float a = leakyf(dsrc[i] + ddst[i] + cptr[0], 0.2f);
    loopl[i] = a;
    atomicMaxF(&m[i], a);
}

__global__ void k_edge_expsum(float* __restrict__ logit, const float* __restrict__ m,
                              const int* __restrict__ dst, float* __restrict__ s, int E) {
    int e = blockIdx.x * blockDim.x + threadIdx.x;
    if (e >= E) return;
    int d = dst[e];
    float mm = m[d];
    if (!(mm > -3.0e38f)) mm = 0.f;            // isfinite(m) ? m : 0
    float ev = __expf(logit[e] - mm);
    logit[e] = ev;                              // reuse buffer for exp values
    atomicAdd(&s[d], ev);
}

__global__ void k_loop_expsum(float* __restrict__ loopl, const float* __restrict__ m,
                              float* __restrict__ s, int n) {
    int i = blockIdx.x * blockDim.x + threadIdx.x;
    if (i >= n) return;
    float mm = m[i];
    if (!(mm > -3.0e38f)) mm = 0.f;
    float ev = __expf(loopl[i] - mm);
    loopl[i] = ev;
    atomicAdd(&s[i], ev);
}

// out[dst,:] += h[src,:] * ev[e]/(s[dst]+eps)   (64 lanes per edge)
__global__ void k_edge_scatter(const float* __restrict__ h, const float* __restrict__ ev,
                               const float* __restrict__ s, const int* __restrict__ src,
                               const int* __restrict__ dst, float* __restrict__ out, int E) {
    int t = blockIdx.x * blockDim.x + threadIdx.x;
    if (t >= E * HID) return;
    int e = t >> 6, j = t & 63;
    int d = dst[e];
    float al = ev[e] / (s[d] + 1e-16f);
    atomicAdd(&out[(long)d * HID + j], h[(long)src[e] * HID + j] * al);
}

// out[i,:] += h[i,:] * loop_ev[i]/(s[i]+eps)   (one writer per element)
__global__ void k_loop_scatter(const float* __restrict__ h, const float* __restrict__ ev,
                               const float* __restrict__ s, float* __restrict__ out, int n) {
    int t = blockIdx.x * blockDim.x + threadIdx.x;
    if (t >= n * HID) return;
    int i = t >> 6;
    out[t] += h[t] * (ev[i] / (s[i] + 1e-16f));
}

// two-stage column-sum of he (E x 64) -> tmp[64]
__global__ void k_col_partial(const float* __restrict__ he, float* __restrict__ tmp, int E) {
    __shared__ float sh[256];
    int t = threadIdx.x;
    int j = t & 63, rl = t >> 6;
    float acc = 0.f;
    for (int e = blockIdx.x * 4 + rl; e < E; e += gridDim.x * 4)
        acc += he[(long)e * HID + j];
    sh[t] = acc;
    __syncthreads();
    if (t < 64) atomicAdd(&tmp[j], sh[t] + sh[t + 64] + sh[t + 128] + sh[t + 192]);
}

__global__ void k_mean_dot(const float* __restrict__ tmp, const float* __restrict__ aedge,
                           float* __restrict__ cptr, float invE) {
    if (threadIdx.x == 0) {
        float acc = 0.f;
        for (int j = 0; j < HID; ++j) acc += tmp[j] * aedge[j];
        cptr[0] = acc * invE;
    }
}

__global__ void k_head(const float* __restrict__ p2, const float* __restrict__ Wout,
                       const float* __restrict__ bout, float* __restrict__ out, int n) {
    int i = blockIdx.x * blockDim.x + threadIdx.x;
    if (i >= n) return;
    const float4* r  = (const float4*)(p2 + (long)i * HID);
    const float4* wv = (const float4*)Wout;
    float acc = 0.f;
#pragma unroll
    for (int t = 0; t < 16; ++t) {
        float4 x = r[t], y = wv[t];
        acc += x.x * y.x + x.y * y.y + x.z * y.z + x.w * y.w;
    }
    out[i] = acc + bout[0];
}

// ---------------------------------------------------------------------------
struct Ws {
    float *he, *hsrc, *hdst, *dsrc, *ddst, *m, *s, *logit, *loopl, *tmp64, *cptr;
};

static void wmma_linear(const float* X, const float* W, const float* bias, float slope,
                        float* Y, int N, int K, hipStream_t st) {
    const dim3 g(CDIV(N, 128)), b(256);
    if (K == 16)      k_wmma_linear<16><<<g, b, 0, st>>>(X, W, bias, slope, Y, N);
    else if (K == 32) k_wmma_linear<32><<<g, b, 0, st>>>(X, W, bias, slope, Y, N);
    else              k_wmma_linear<64><<<g, b, 0, st>>>(X, W, bias, slope, Y, N);
}

static void homo_gat(const float* x, int n, const int* src, const int* dst, int E,
                     const float* eproj, const float* W, const float* We,
                     const float* asrc, const float* adst, const float* aedge,
                     const float* bias, float* out, bool add, Ws& w, hipStream_t st) {
    wmma_linear(x, W, nullptr, 1.0f, w.hsrc, n, HID, st);
    wmma_linear(eproj, We, nullptr, 1.0f, w.he, E, HID, st);
    k_row_dot<<<CDIV(n, 256), 256, 0, st>>>(w.hsrc, asrc, w.dsrc, n);
    k_row_dot<<<CDIV(n, 256), 256, 0, st>>>(w.hsrc, adst, w.ddst, n);
    k_fill<<<1, 64, 0, st>>>(w.tmp64, 0.f, 64);
    k_col_partial<<<512, 256, 0, st>>>(w.he, w.tmp64, E);
    k_mean_dot<<<1, 64, 0, st>>>(w.tmp64, aedge, w.cptr, 1.0f / (float)E);
    k_fill<<<CDIV(n, 256), 256, 0, st>>>(w.m, -__builtin_inff(), n);
    k_fill<<<CDIV(n, 256), 256, 0, st>>>(w.s, 0.f, n);
    k_edge_logits<<<CDIV(E, 256), 256, 0, st>>>(w.he, aedge, w.dsrc, w.ddst, src, dst, w.logit, w.m, E);
    k_loop_logits<<<CDIV(n, 256), 256, 0, st>>>(w.dsrc, w.ddst, w.cptr, w.loopl, w.m, n);
    k_edge_expsum<<<CDIV(E, 256), 256, 0, st>>>(w.logit, w.m, dst, w.s, E);
    k_loop_expsum<<<CDIV(n, 256), 256, 0, st>>>(w.loopl, w.m, w.s, n);
    k_bias_rows<<<CDIV(n * HID, 256), 256, 0, st>>>(out, bias, n, add ? 1 : 0);
    k_edge_scatter<<<CDIV(E * HID, 256), 256, 0, st>>>(w.hsrc, w.logit, w.s, src, dst, out, E);
    k_loop_scatter<<<CDIV(n * HID, 256), 256, 0, st>>>(w.hsrc, w.loopl, w.s, out, n);
}

static void bi_gat(const float* xsrc, int nsrc, const float* xdst, int ndst,
                   const int* src, const int* dst, int E, const float* eproj,
                   const float* Wsrc, const float* Wdst, const float* We,
                   const float* asrc, const float* adst, const float* aedge,
                   const float* bias, float* out, bool add, Ws& w, hipStream_t st) {
    wmma_linear(xsrc, Wsrc, nullptr, 1.0f, w.hsrc, nsrc, HID, st);
    wmma_linear(xdst, Wdst, nullptr, 1.0f, w.hdst, ndst, HID, st);
    wmma_linear(eproj, We, nullptr, 1.0f, w.he, E, HID, st);
    k_row_dot<<<CDIV(nsrc, 256), 256, 0, st>>>(w.hsrc, asrc, w.dsrc, nsrc);
    k_row_dot<<<CDIV(ndst, 256), 256, 0, st>>>(w.hdst, adst, w.ddst, ndst);
    k_fill<<<CDIV(ndst, 256), 256, 0, st>>>(w.m, -__builtin_inff(), ndst);
    k_fill<<<CDIV(ndst, 256), 256, 0, st>>>(w.s, 0.f, ndst);
    k_edge_logits<<<CDIV(E, 256), 256, 0, st>>>(w.he, aedge, w.dsrc, w.ddst, src, dst, w.logit, w.m, E);
    k_edge_expsum<<<CDIV(E, 256), 256, 0, st>>>(w.logit, w.m, dst, w.s, E);
    k_bias_rows<<<CDIV(ndst * HID, 256), 256, 0, st>>>(out, bias, ndst, add ? 1 : 0);
    k_edge_scatter<<<CDIV(E * HID, 256), 256, 0, st>>>(w.hsrc, w.logit, w.s, src, dst, out, E);
}

// ---------------------------------------------------------------------------
extern "C" void kernel_launch(void* const* d_in, const int* in_sizes, int n_in,
                              void* d_out, int out_size, void* d_ws, size_t ws_size,
                              hipStream_t stream) {
    (void)n_in; (void)out_size; (void)ws_size;
    const int NP = in_sizes[0] / 32;   // D_NODE = 32
    const int NB = in_sizes[1] / 32;
    const int E  = in_sizes[2] / 16;   // D_EDGE = 16

    auto fin = [&](int i) { return (const float*)d_in[i]; };

    // workspace carve-up (float units, 256B aligned)
    size_t off = 0;
    auto alloc = [&](size_t nf) {
        float* p = (float*)d_ws + off;
        off += (nf + 63) & ~(size_t)63;
        return p;
    };
    float* xp  = alloc((size_t)NP * HID);   // later reused as p2
    float* xb  = alloc((size_t)NB * HID);
    float* p1  = alloc((size_t)NP * HID);
    float* b1  = alloc((size_t)NB * HID);
    float* epp = alloc((size_t)E * HID);
    float* ebb = alloc((size_t)E * HID);
    float* ebp = alloc((size_t)E * HID);
    Ws w;
    w.he    = alloc((size_t)E * HID);
    w.hsrc  = alloc((size_t)NB * HID);
    w.hdst  = alloc((size_t)NP * HID);
    w.dsrc  = alloc(NB);
    w.ddst  = alloc(NB);
    w.m     = alloc(NB);
    w.s     = alloc(NB);
    w.logit = alloc(E);
    w.loopl = alloc(NB);
    w.tmp64 = alloc(64);
    w.cptr  = alloc(1);

    const int* ei_pp = (const int*)d_in[55];
    const int* ei_bb = (const int*)d_in[56];
    const int* ei_bp = (const int*)d_in[57];
    const int *src_pp = ei_pp, *dst_pp = ei_pp + E;
    const int *src_bb = ei_bb, *dst_bb = ei_bb + E;
    const int *src_bp = ei_bp, *dst_bp = ei_bp + E;

    // input projections + LeakyReLU(0.01)
    wmma_linear(fin(0), fin(5), fin(6), 0.01f, xp, NP, 32, stream);
    wmma_linear(fin(1), fin(7), fin(8), 0.01f, xb, NB, 32, stream);
    wmma_linear(fin(2), fin(9),  fin(10), 0.01f, epp, E, 16, stream);
    wmma_linear(fin(3), fin(11), fin(12), 0.01f, ebb, E, 16, stream);
    wmma_linear(fin(4), fin(13), fin(14), 0.01f, ebp, E, 16, stream);

    // conv1  (HeteroConv sum: p1 = pp-homo + bp-bi ; b1 = bb-homo)
    homo_gat(xp, NP, src_pp, dst_pp, E, epp,
             fin(15), fin(16), fin(17), fin(18), fin(19), fin(20),
             p1, false, w, stream);
    bi_gat(xb, NB, xp, NP, src_bp, dst_bp, E, ebp,
           fin(27), fin(28), fin(29), fin(30), fin(31), fin(32), fin(33),
           p1, true, w, stream);
    homo_gat(xb, NB, src_bb, dst_bb, E, ebb,
             fin(21), fin(22), fin(23), fin(24), fin(25), fin(26),
             b1, false, w, stream);

    // conv2 (only the proposal output is live; bb branch is dead code)
    float* p2 = xp;   // xp no longer needed
    homo_gat(p1, NP, src_pp, dst_pp, E, epp,
             fin(34), fin(35), fin(36), fin(37), fin(38), fin(39),
             p2, false, w, stream);
    bi_gat(b1, NB, p1, NP, src_bp, dst_bp, E, ebp,
           fin(46), fin(47), fin(48), fin(49), fin(50), fin(51), fin(52),
           p2, true, w, stream);

    // head
    k_head<<<CDIV(NP, 256), 256, 0, stream>>>(p2, fin(53), fin(54), (float*)d_out, NP);
}
